// GroupedQueryAttention_62740882260309
// MI455X (gfx1250) — matrix operand order repaired and checked by reading, not compiled, not run on hardware
//
#include <hip/hip_runtime.h>
#include <hip/hip_bf16.h>

// ---------------- problem constants ----------------
constexpr int Bb   = 2;
constexpr int Ll   = 2048;
constexpr int Dd   = 2048;
constexpr int Hh   = 16;
constexpr int KVHh = 4;
constexpr int HDd  = 128;          // head dim
constexpr int NREP = Hh / KVHh;    // 4
constexpr float SCALE = 0.08838834764831845f;   // 128^-0.5

typedef __attribute__((ext_vector_type(16))) __bf16 v16bf;
typedef __attribute__((ext_vector_type(8)))  float  v8f;

union ABu { uint4 q[2]; v16bf v; };

__device__ __forceinline__ unsigned short f2bf(float f) {
    unsigned u = __float_as_uint(f);
    u += 0x7FFFu + ((u >> 16) & 1u);          // round-to-nearest-even
    return (unsigned short)(u >> 16);
}

// ---- CDNA5 async global->LDS copy of 16 bytes (tracked by ASYNCcnt) ----
__device__ __forceinline__ void async_cp16(unsigned short* lds_dst, const unsigned short* gsrc) {
    unsigned lds = (unsigned)(unsigned long long)lds_dst;      // low 32 bits of shared ptr = LDS offset
    unsigned long long ga = (unsigned long long)gsrc;
    asm volatile("global_load_async_to_lds_b128 %0, %1, off"
                 :: "v"(lds), "v"(ga) : "memory");
}
__device__ __forceinline__ void wait_async0() {
    asm volatile("s_wait_asynccnt 0x0" ::: "memory");
}

// -------- A-operand staging: bf16 -> async DMA; f32 -> convert + ds_store --------
__device__ __forceinline__ void stageA8(unsigned short* dst, const unsigned short* src) {
    async_cp16(dst, src);
}
__device__ __forceinline__ void stageA8(unsigned short* dst, const float* src) {
    float4 a = ((const float4*)src)[0];
    float4 b = ((const float4*)src)[1];
    uint4 o;
    o.x = (unsigned)f2bf(a.x) | ((unsigned)f2bf(a.y) << 16);
    o.y = (unsigned)f2bf(a.z) | ((unsigned)f2bf(a.w) << 16);
    o.z = (unsigned)f2bf(b.x) | ((unsigned)f2bf(b.y) << 16);
    o.w = (unsigned)f2bf(b.z) | ((unsigned)f2bf(b.w) << 16);
    *(uint4*)dst = o;
}

// =====================================================================
// Generic batched NT GEMM:  C[m,n] = sum_k A[m,k] * Bt[n,k]
// A: bf16(raw ushort, async-DMA staged) or f32 (converted during staging);
// Bt: bf16 (always async-DMA staged).
// Block tile 128x64, 8 waves, each wave 2x2 WMMA f32_16x16x32_bf16 tiles,
// double-buffered LDS, one barrier per K-iteration.
// Batch (z) offsets: zo = z/gqaH, zi = z%gqaH;
//   A  += zo*aOut + zi*aIn ; Bt += zo*bOut + (zi/nRep)*bIn ; C += zo*cOut + zi*cIn
// All dims are multiples of the tile sizes for every launch here.
// =====================================================================
template <typename AT>
__global__ __launch_bounds__(256) void gemm_nt_wmma(
    const AT* __restrict__ A, const unsigned short* __restrict__ Bt,
    float* __restrict__ C,
    int Kd, int lda, int ldb, int ldc,
    long long aOut, long long aIn, long long bOut, long long bIn,
    long long cOut, long long cIn, int gqaH, int nRep)
{
    constexpr int SA = 48;   // LDS row stride (bf16 elems), 96 B (16B aligned, bank-spread)
    constexpr int SB = 48;
    __shared__ unsigned short As[2][128 * SA];
    __shared__ unsigned short Bs[2][64 * SB];

    const int z  = blockIdx.z;
    const int zo = z / gqaH, zi = z % gqaH;
    A  += zo * aOut + (long long)zi * aIn;
    Bt += zo * bOut + (long long)(zi / nRep) * bIn;
    C  += zo * cOut + (long long)zi * cIn;

    const int m0 = blockIdx.x * 128;
    const int n0 = blockIdx.y * 64;
    const int tid  = threadIdx.x;
    const int lane = tid & 31, wid = tid >> 5;
    const int wm = (wid & 3) * 32;      // wave row offset (4 waves down)
    const int wn = (wid >> 2) * 32;     // wave col offset (2 waves across)
    const int lr = lane & 15, lh = lane >> 4;

    // per-thread staging coordinates (fixed across iterations)
    const int arow0 = tid >> 2;                 // A chunk 0: rows 0..63
    const int arow1 = (tid + 256) >> 2;         // A chunk 1: rows 64..127
    const int ak8   = (tid & 3) * 8;
    const int brow  = tid >> 2;                 // B rows 0..63
    const int bk8   = (tid & 3) * 8;

    v8f zero = {0.f,0.f,0.f,0.f,0.f,0.f,0.f,0.f};
    v8f acc[2][2] = {{zero, zero}, {zero, zero}};

    const int nIter = Kd >> 5;

    auto issue = [&](int buf, int k0) {
        stageA8(&As[buf][arow0 * SA + ak8], A + (long long)(m0 + arow0) * lda + k0 + ak8);
        stageA8(&As[buf][arow1 * SA + ak8], A + (long long)(m0 + arow1) * lda + k0 + ak8);
        async_cp16(&Bs[buf][brow * SB + bk8], Bt + (long long)(n0 + brow) * ldb + k0 + bk8);
    };

    issue(0, 0);

    for (int it = 0; it < nIter; ++it) {
        const int cur = it & 1;
        wait_async0();                 // my async copies for `cur` have landed in LDS
        __syncthreads();               // everyone's landed; everyone done reading `cur^1`
        if (it + 1 < nIter) issue(cur ^ 1, (it + 1) << 5);

        // ---- build operands per documented VGPR layouts ----
        v16bf a[2], b[2];
        #pragma unroll
        for (int mi = 0; mi < 2; ++mi) {
            int row = wm + mi * 16 + lr;                 // M = lane%16
            ABu u;
            u.q[0] = *(const uint4*)&As[cur][row * SA + 8 * lh];        // K = 8*half + 0..7
            u.q[1] = *(const uint4*)&As[cur][row * SA + 16 + 8 * lh];   // K = 16 + 8*half + 0..7
            a[mi] = u.v;
        }
        #pragma unroll
        for (int ni = 0; ni < 2; ++ni) {
            int n = wn + ni * 16 + lr;                   // N = lane%16
            ABu u;
            u.q[0] = *(const uint4*)&Bs[cur][n * SB + 8 * lh];          // K = 8*half + 0..7
            u.q[1] = *(const uint4*)&Bs[cur][n * SB + 16 + 8 * lh];     // K = 16 + 8*half + 0..7
            b[ni] = u.v;
        }
        #pragma unroll
        for (int mi = 0; mi < 2; ++mi)
            #pragma unroll
            for (int ni = 0; ni < 2; ++ni)
                acc[mi][ni] = __builtin_amdgcn_wmma_f32_16x16x32_bf16(
                    false, a[mi], false, b[ni], (short)0, acc[mi][ni], false, false);
    }

    // ---- store C (non-temporal; big outputs don't fit L2 anyway) ----
    #pragma unroll
    for (int mi = 0; mi < 2; ++mi)
        #pragma unroll
        for (int ni = 0; ni < 2; ++ni) {
            int rbase = m0 + wm + mi * 16 + lh * 8;
            int col   = n0 + wn + ni * 16 + lr;
            float* Cp = C + (long long)rbase * ldc + col;
            #pragma unroll
            for (int r = 0; r < 8; ++r) {
                __builtin_nontemporal_store(acc[mi][ni][r], Cp);
                Cp += ldc;
            }
        }
}

// ---------------- elementwise f32 -> bf16 ----------------
__global__ void cvt_bf16(const float* __restrict__ in, unsigned short* __restrict__ out, int n) {
    for (int i = blockIdx.x * blockDim.x + threadIdx.x; i < n; i += gridDim.x * blockDim.x)
        out[i] = f2bf(in[i]);
}

// ------------- transpose + convert: W[k,n] (f32) -> Wt[n,k] (bf16) -------------
__global__ void tcvt_bf16(const float* __restrict__ W, unsigned short* __restrict__ Wt,
                          int Kd, int N) {
    long long total = (long long)Kd * N;
    for (long long i = blockIdx.x * blockDim.x + threadIdx.x; i < total;
         i += (long long)gridDim.x * blockDim.x) {
        int n = (int)(i / Kd), k = (int)(i % Kd);
        Wt[(long long)n * Kd + k] = f2bf(W[(long long)k * N + n]);
    }
}

// -------- RoPE + scale + head packing + V transpose --------
__global__ __launch_bounds__(128) void rope_pack(
    const float* __restrict__ Qf, const float* __restrict__ Kf, const float* __restrict__ Vf,
    const float* __restrict__ cosT, const float* __restrict__ sinT,
    const int* __restrict__ pos,
    unsigned short* __restrict__ Qh, unsigned short* __restrict__ Kh,
    unsigned short* __restrict__ Vt)
{
    const int t = blockIdx.x;              // token index 0..B*L-1
    const int d = threadIdx.x;             // 0..127
    const int b = t / Ll, l = t % Ll;
    const int p = pos[t];
    const float c = cosT[(long long)p * HDd + d];
    const float s = sinT[(long long)p * HDd + d];

    #pragma unroll 4
    for (int h = 0; h < Hh; ++h) {
        long long base = (long long)t * Dd + h * HDd;
        float q0 = Qf[base + d];
        float qr = (d < 64) ? -Qf[base + d + 64] : Qf[base + d - 64];
        Qh[((long long)(b * Hh + h) * Ll + l) * HDd + d] = f2bf((q0 * c + qr * s) * SCALE);
    }
    #pragma unroll
    for (int h = 0; h < KVHh; ++h) {
        long long base = (long long)t * (KVHh * HDd) + h * HDd;
        float k0 = Kf[base + d];
        float kr = (d < 64) ? -Kf[base + d + 64] : Kf[base + d - 64];
        Kh[((long long)(b * KVHh + h) * Ll + l) * HDd + d] = f2bf(k0 * c + kr * s);
        float vv = Vf[base + d];
        Vt[((long long)(b * KVHh + h) * HDd + d) * Ll + l] = f2bf(vv);
    }
}

// -------- in-place row softmax over L=2048 columns --------
__global__ __launch_bounds__(256) void softmax_rows(float* __restrict__ w) {
    __shared__ float red[256];
    float* p = w + (long long)blockIdx.x * Ll;
    const int tid = threadIdx.x;

    float v[8], mx = -3.4e38f;
    #pragma unroll
    for (int i = 0; i < 8; ++i) { v[i] = p[tid + 256 * i]; mx = fmaxf(mx, v[i]); }
    red[tid] = mx; __syncthreads();
    for (int s = 128; s > 0; s >>= 1) { if (tid < s) red[tid] = fmaxf(red[tid], red[tid + s]); __syncthreads(); }
    mx = red[0]; __syncthreads();

    float sum = 0.f;
    #pragma unroll
    for (int i = 0; i < 8; ++i) { v[i] = __expf(v[i] - mx); sum += v[i]; }
    red[tid] = sum; __syncthreads();
    for (int s = 128; s > 0; s >>= 1) { if (tid < s) red[tid] += red[tid + s]; __syncthreads(); }
    float inv = 1.f / red[0];
    #pragma unroll
    for (int i = 0; i < 8; ++i) p[tid + 256 * i] = v[i] * inv;
}

// =====================================================================
extern "C" void kernel_launch(void* const* d_in, const int* in_sizes, int n_in,
                              void* d_out, int out_size, void* d_ws, size_t ws_size,
                              hipStream_t stream) {
    (void)in_sizes; (void)n_in; (void)out_size; (void)ws_size;
    const float* x    = (const float*)d_in[0];
    const float* cosT = (const float*)d_in[1];
    const float* sinT = (const float*)d_in[2];
    const int*   pos  = (const int*)d_in[3];
    const float* Wq   = (const float*)d_in[4];
    const float* Wk   = (const float*)d_in[5];
    const float* Wv   = (const float*)d_in[6];
    const float* Wo   = (const float*)d_in[7];

    float* out_attn = (float*)d_out;                                    // [B,L,D]
    float* out_w    = out_attn + (long long)Bb * Ll * Dd;               // [B,H,L,L]

    const int BL = Bb * Ll;                 // 4096 tokens
    char* ws = (char*)d_ws;
    size_t off = 0;
    auto take = [&](size_t bytes) { char* p = ws + off; off += bytes; return p; };
    unsigned short* x_bf  = (unsigned short*)take((size_t)BL * Dd * 2);
    unsigned short* Wq_t  = (unsigned short*)take((size_t)Dd * Dd * 2);
    unsigned short* Wk_t  = (unsigned short*)take((size_t)Dd * (KVHh*HDd) * 2);
    unsigned short* Wv_t  = (unsigned short*)take((size_t)Dd * (KVHh*HDd) * 2);
    unsigned short* Wo_t  = (unsigned short*)take((size_t)Dd * Dd * 2);
    float*          Qf    = (float*)take((size_t)BL * Dd * 4);
    float*          Kf    = (float*)take((size_t)BL * (KVHh*HDd) * 4);
    float*          Vf    = (float*)take((size_t)BL * (KVHh*HDd) * 4);
    unsigned short* Qh    = (unsigned short*)take((size_t)BL * Dd * 2);
    unsigned short* Kh    = (unsigned short*)take((size_t)BL * (KVHh*HDd) * 2);
    unsigned short* Vt    = (unsigned short*)take((size_t)BL * (KVHh*HDd) * 2);
    float*          attn  = (float*)take((size_t)BL * Dd * 4);

    // 1) convert activations & transpose-convert weights to bf16
    cvt_bf16<<<2048, 256, 0, stream>>>(x, x_bf, BL * Dd);
    tcvt_bf16<<<2048, 256, 0, stream>>>(Wq, Wq_t, Dd, Dd);
    tcvt_bf16<<<1024, 256, 0, stream>>>(Wk, Wk_t, Dd, KVHh * HDd);
    tcvt_bf16<<<1024, 256, 0, stream>>>(Wv, Wv_t, Dd, KVHh * HDd);
    tcvt_bf16<<<2048, 256, 0, stream>>>(Wo, Wo_t, Dd, Dd);

    // 2) projections: Qf = x@Wq, Kf = x@Wk, Vf = x@Wv   (NT vs transposed weights)
    gemm_nt_wmma<unsigned short><<<dim3(BL/128, Dd/64, 1), 256, 0, stream>>>(
        x_bf, Wq_t, Qf, Dd, Dd, Dd, Dd, 0,0,0,0,0,0, 1, 1);
    gemm_nt_wmma<unsigned short><<<dim3(BL/128, (KVHh*HDd)/64, 1), 256, 0, stream>>>(
        x_bf, Wk_t, Kf, Dd, Dd, Dd, KVHh*HDd, 0,0,0,0,0,0, 1, 1);
    gemm_nt_wmma<unsigned short><<<dim3(BL/128, (KVHh*HDd)/64, 1), 256, 0, stream>>>(
        x_bf, Wv_t, Vf, Dd, Dd, Dd, KVHh*HDd, 0,0,0,0,0,0, 1, 1);

    // 3) RoPE + scale + pack heads (+ V^T)
    rope_pack<<<BL, 128, 0, stream>>>(Qf, Kf, Vf, cosT, sinT, pos, Qh, Kh, Vt);

    // 4) scores: S[b,h] = Qh[b,h] @ Kh[b,h/4]^T  -> d_out weights region (f32)
    {
        long long aOut = (long long)Hh * Ll * HDd, aIn = (long long)Ll * HDd;
        long long bOut = (long long)KVHh * Ll * HDd, bIn = (long long)Ll * HDd;
        long long cOut = (long long)Hh * Ll * Ll,  cIn = (long long)Ll * Ll;
        gemm_nt_wmma<unsigned short><<<dim3(Ll/128, Ll/64, Bb*Hh), 256, 0, stream>>>(
            Qh, Kh, out_w, HDd, HDd, HDd, Ll,
            aOut, aIn, bOut, bIn, cOut, cIn, Hh, NREP);
    }

    // 5) softmax in place on the weights output
    softmax_rows<<<Bb * Hh * Ll, 256, 0, stream>>>(out_w);

    // 6) attn[b,l,h*HD+:] = P[b,h] @ V[b,h/4]    (A is f32, converted in staging)
    {
        long long aOut = (long long)Hh * Ll * Ll, aIn = (long long)Ll * Ll;
        long long bOut = (long long)KVHh * HDd * Ll, bIn = (long long)HDd * Ll;
        long long cOut = (long long)Ll * Dd, cIn = (long long)HDd;
        gemm_nt_wmma<float><<<dim3(Ll/128, HDd/64, Bb*Hh), 256, 0, stream>>>(
            out_w, Vt, attn, Ll, Ll, Ll, Dd,
            aOut, aIn, bOut, bIn, cOut, cIn, Hh, NREP);
    }

    // 7) output projection: out = attn @ Wo
    gemm_nt_wmma<float><<<dim3(BL/128, Dd/64, 1), 256, 0, stream>>>(
        attn, Wo_t, out_attn, Dd, Dd, Dd, Dd, 0,0,0,0,0,0, 1, 1);
}
